// TrafficGNN_20237885899322
// MI455X (gfx1250) — matrix operand order repaired and checked by reading, not compiled, not run on hardware
//
#include <hip/hip_runtime.h>
#include <hip/hip_bf16.h>

// ---------------------------------------------------------------------------
// TrafficGNN for MI455X (gfx1250): 2x GCNConv + 2x GATConv + pool + MLP.
// Dense node transforms: v_wmma_f32_16x16x32_bf16, 4 WMMAs per wave per
// k-step (one A fragment reused across 4 column tiles), B fragments staged
// transposed in LDS for ds_load_b128 access. Edge scatters are wave-per-edge
// float atomics (memory/L2 bound phase).
// ---------------------------------------------------------------------------

typedef __attribute__((ext_vector_type(16))) __bf16 v16bf;
typedef __attribute__((ext_vector_type(8)))  float  v8f;

#define HIDF 64

// Order-preserving float -> uint encoding for atomicMax-based segment max.
__device__ __forceinline__ unsigned fenc(float f) {
  unsigned u = __float_as_uint(f);
  return (u & 0x80000000u) ? ~u : (u | 0x80000000u);
}
__device__ __forceinline__ float fdec(unsigned e) {
  unsigned u = (e & 0x80000000u) ? (e & 0x7fffffffu) : ~e;
  return __uint_as_float(u);
}
#define NEG_INF_ENC 0x007FFFFFu  // fenc(-inf)

// ---------------------------------------------------------------------------
// WMMA GEMM: C[M x 64] = A[M x K] * W[K x 64], A/W f32 row-major, K in {32,64}.
// Block = 256 threads = 8 waves; each wave computes a 16-row x 64-col slab:
// one 16x32 bf16 A fragment reused by 4 WMMAs against 4 B column tiles.
// W is staged to LDS transposed as bf16 [col][K] with padded stride so each
// lane's B fragment (16 contiguous K values) is two 16B ds loads,
// conflict-free across the 16 active columns.
// ---------------------------------------------------------------------------
__global__ void __launch_bounds__(256)
gemm_wmma_bf16(const float* __restrict__ A, const float* __restrict__ W,
               float* __restrict__ C, int M, int K) {
  __shared__ __bf16 wlt[HIDF * 72];  // max K=64 -> stride 72
  const int KS = K + 8;              // padded K stride (16B aligned, bank-spread)
  for (int i = threadIdx.x; i < K * HIDF; i += 256) {
    const int k = i >> 6, n = i & 63;        // W[k][n] row-major
    wlt[n * KS + k] = (__bf16)W[i];
  }
  __syncthreads();

  const int wave = threadIdx.x >> 5;
  const int lane = threadIdx.x & 31;
  const int row0 = (blockIdx.x * 8 + wave) * 16;
  const int colb = lane & 15;   // column within each 16-col tile
  const int hi   = lane >> 4;   // lane half selects K sub-range

  int arow = row0 + (lane & 15);
  if (arow >= M) arow = M - 1;  // clamp: OOB rows computed but never stored
  const float* __restrict__ ap = A + (size_t)arow * K + hi * 16;

  v8f c0 = {}, c1 = {}, c2 = {}, c3 = {};
  for (int k0 = 0; k0 < K; k0 += 32) {
    // A fragment: contiguous K order (VGPR j: K=2j,2j+1; hi half: +16), matching B
    v16bf a;
    #pragma unroll
    for (int j = 0; j < 8; ++j) {
      const int kk = 2 * j;
      float2 v = *(const float2*)(ap + k0 + kk);
      a[2 * j]     = (__bf16)v.x;
      a[2 * j + 1] = (__bf16)v.y;
    }
    // B fragments: lanes 0-15 hold K=kb..kb+15 (N=colb), lanes 16-31 K+16.
    const int kb = k0 + hi * 16;
    union Frag { uint4 q[2]; v16bf v; };
    Frag b0, b1, b2, b3;
    {
      const __bf16* p0 = wlt + (0 * 16 + colb) * KS + kb;
      const __bf16* p1 = wlt + (1 * 16 + colb) * KS + kb;
      const __bf16* p2 = wlt + (2 * 16 + colb) * KS + kb;
      const __bf16* p3 = wlt + (3 * 16 + colb) * KS + kb;
      b0.q[0] = *(const uint4*)(p0); b0.q[1] = *(const uint4*)(p0 + 8);
      b1.q[0] = *(const uint4*)(p1); b1.q[1] = *(const uint4*)(p1 + 8);
      b2.q[0] = *(const uint4*)(p2); b2.q[1] = *(const uint4*)(p2 + 8);
      b3.q[0] = *(const uint4*)(p3); b3.q[1] = *(const uint4*)(p3 + 8);
    }
    c0 = __builtin_amdgcn_wmma_f32_16x16x32_bf16(false, a, false, b0.v, (short)0, c0, false, false);
    c1 = __builtin_amdgcn_wmma_f32_16x16x32_bf16(false, a, false, b1.v, (short)0, c1, false, false);
    c2 = __builtin_amdgcn_wmma_f32_16x16x32_bf16(false, a, false, b2.v, (short)0, c2, false, false);
    c3 = __builtin_amdgcn_wmma_f32_16x16x32_bf16(false, a, false, b3.v, (short)0, c3, false, false);
  }

  // C layout: VGPR i -> M = i + 8*hi. Wave-uniform full/partial branch.
  const int mbase = row0 + hi * 8;
  if (row0 + 16 <= M) {
    #pragma unroll
    for (int i = 0; i < 8; ++i) {
      float* cr = C + (size_t)(mbase + i) * HIDF + colb;
      cr[0]  = c0[i];
      cr[16] = c1[i];
      cr[32] = c2[i];
      cr[48] = c3[i];
    }
  } else {
    #pragma unroll
    for (int i = 0; i < 8; ++i) {
      if (mbase + i < M) {
        float* cr = C + (size_t)(mbase + i) * HIDF + colb;
        cr[0]  = c0[i];
        cr[16] = c1[i];
        cr[32] = c2[i];
        cr[48] = c3[i];
      }
    }
  }
}

// ---------------------------------------------------------------------------
// Utility fills
// ---------------------------------------------------------------------------
__global__ void fill_f32(float* __restrict__ p, float v, int n) {
  int i = blockIdx.x * blockDim.x + threadIdx.x;
  if (i < n) p[i] = v;
}
__global__ void fill_u32(unsigned* __restrict__ p, unsigned v, int n) {
  int i = blockIdx.x * blockDim.x + threadIdx.x;
  if (i < n) p[i] = v;
}

// ---------------------------------------------------------------------------
// Degree / normalization (self loops folded in by initializing deg = 1)
// ---------------------------------------------------------------------------
__global__ void deg_count(const int* __restrict__ dst, float* __restrict__ deg, int E) {
  int e = blockIdx.x * blockDim.x + threadIdx.x;
  if (e < E) atomicAdd(&deg[dst[e]], 1.0f);
}
__global__ void to_rsqrt(float* __restrict__ d, int n) {
  int i = blockIdx.x * blockDim.x + threadIdx.x;
  if (i < n) d[i] = rsqrtf(d[i]);  // deg >= 1 always (self loop)
}

// ---------------------------------------------------------------------------
// GCN: out[dst] += dinv[src]*dinv[dst] * h[src]; wave per edge, 2 feats/lane
// ---------------------------------------------------------------------------
__global__ void gcn_scatter(const int* __restrict__ src, const int* __restrict__ dst,
                            const float* __restrict__ dinv, const float* __restrict__ h,
                            float* __restrict__ out, int E) {
  int t = blockIdx.x * blockDim.x + threadIdx.x;
  int e = t >> 5;
  if (e >= E) return;
  int lane = t & 31;
  int s = src[e], d = dst[e];
  float w = dinv[s] * dinv[d];
  float2 hv = *(const float2*)(h + (size_t)s * HIDF + lane * 2);
  atomicAdd(out + (size_t)d * HIDF + lane * 2,     w * hv.x);
  atomicAdd(out + (size_t)d * HIDF + lane * 2 + 1, w * hv.y);
}
// out = relu(out + dinv^2 * h (self loop) + bias)
__global__ void gcn_finalize(float* __restrict__ out, const float* __restrict__ h,
                             const float* __restrict__ dinv, const float* __restrict__ bias,
                             int n) {
  int i = blockIdx.x * blockDim.x + threadIdx.x;
  if (i >= n * HIDF) return;
  int node = i >> 6, f = i & 63;
  float di = dinv[node];
  out[i] = fmaxf(out[i] + di * di * h[i] + bias[f], 0.0f);
}

// ---------------------------------------------------------------------------
// GAT: per-node scores, segment-max softmax over incoming edges, weighted sum
// ---------------------------------------------------------------------------
__global__ void gat_scores(const float* __restrict__ h, const float* __restrict__ as,
                           const float* __restrict__ ad, float* __restrict__ es,
                           float* __restrict__ ed, int n) {
  int t = blockIdx.x * blockDim.x + threadIdx.x;
  int node = t >> 5;
  if (node >= n) return;
  int lane = t & 31;
  float2 hv = *(const float2*)(h + (size_t)node * HIDF + lane * 2);
  float2 s2 = *(const float2*)(as + lane * 2);
  float2 d2 = *(const float2*)(ad + lane * 2);
  float vs = hv.x * s2.x + hv.y * s2.y;
  float vd = hv.x * d2.x + hv.y * d2.y;
  #pragma unroll
  for (int o = 16; o > 0; o >>= 1) {
    vs += __shfl_down(vs, o, 32);
    vd += __shfl_down(vd, o, 32);
  }
  if (lane == 0) { es[node] = vs; ed[node] = vd; }
}
__global__ void gat_edge_max(const int* __restrict__ src, const int* __restrict__ dst,
                             const float* __restrict__ es, const float* __restrict__ ed,
                             float* __restrict__ ebuf, unsigned* __restrict__ emax, int E) {
  int e = blockIdx.x * blockDim.x + threadIdx.x;
  if (e >= E) return;
  float v = es[src[e]] + ed[dst[e]];
  v = (v > 0.0f) ? v : 0.2f * v;  // leaky relu
  ebuf[e] = v;
  atomicMax(&emax[dst[e]], fenc(v));
}
__global__ void gat_self_max(const float* __restrict__ es, const float* __restrict__ ed,
                             float* __restrict__ eself, unsigned* __restrict__ emax, int n) {
  int i = blockIdx.x * blockDim.x + threadIdx.x;
  if (i >= n) return;
  float v = es[i] + ed[i];
  v = (v > 0.0f) ? v : 0.2f * v;
  eself[i] = v;
  atomicMax(&emax[i], fenc(v));
}
__global__ void gat_edge_exp(const int* __restrict__ dst, float* __restrict__ ebuf,
                             const unsigned* __restrict__ emax, float* __restrict__ denom, int E) {
  int e = blockIdx.x * blockDim.x + threadIdx.x;
  if (e >= E) return;
  int d = dst[e];
  float x = __expf(ebuf[e] - fdec(emax[d]));
  ebuf[e] = x;
  atomicAdd(&denom[d], x);
}
__global__ void gat_self_exp(float* __restrict__ eself, const unsigned* __restrict__ emax,
                             float* __restrict__ denom, int n) {
  int i = blockIdx.x * blockDim.x + threadIdx.x;
  if (i >= n) return;
  float x = __expf(eself[i] - fdec(emax[i]));
  eself[i] = x;
  atomicAdd(&denom[i], x);
}
__global__ void gat_scatter(const int* __restrict__ src, const int* __restrict__ dst,
                            const float* __restrict__ ebuf, const float* __restrict__ denom,
                            const float* __restrict__ h, float* __restrict__ out, int E) {
  int t = blockIdx.x * blockDim.x + threadIdx.x;
  int e = t >> 5;
  if (e >= E) return;
  int lane = t & 31;
  int s = src[e], d = dst[e];
  float alpha = ebuf[e] / denom[d];
  float2 hv = *(const float2*)(h + (size_t)s * HIDF + lane * 2);
  atomicAdd(out + (size_t)d * HIDF + lane * 2,     alpha * hv.x);
  atomicAdd(out + (size_t)d * HIDF + lane * 2 + 1, alpha * hv.y);
}
__global__ void gat_finalize(float* __restrict__ out, const float* __restrict__ h,
                             const float* __restrict__ eself, const float* __restrict__ denom,
                             const float* __restrict__ bias, int n) {
  int i = blockIdx.x * blockDim.x + threadIdx.x;
  if (i >= n * HIDF) return;
  int node = i >> 6, f = i & 63;
  float alpha = eself[node] / denom[node];
  out[i] = fmaxf(out[i] + alpha * h[i] + bias[f], 0.0f);
}

// ---------------------------------------------------------------------------
// Mean pool + MLP head
// ---------------------------------------------------------------------------
__global__ void pool_sum(const float* __restrict__ h, const int* __restrict__ batch,
                         float* __restrict__ psum, float* __restrict__ pcnt, int n) {
  int t = blockIdx.x * blockDim.x + threadIdx.x;
  int node = t >> 5;
  if (node >= n) return;
  int lane = t & 31;
  int g = batch[node];
  float2 hv = *(const float2*)(h + (size_t)node * HIDF + lane * 2);
  atomicAdd(&psum[g * HIDF + lane * 2],     hv.x);
  atomicAdd(&psum[g * HIDF + lane * 2 + 1], hv.y);
  if (lane == 0) atomicAdd(&pcnt[g], 1.0f);
}
__global__ void mlp_head(const float* __restrict__ psum, const float* __restrict__ pcnt,
                         const float* __restrict__ Wf1, const float* __restrict__ bf1,
                         const float* __restrict__ Wf2, const float* __restrict__ bf2,
                         float* __restrict__ out, int G) {
  int g = blockIdx.x * blockDim.x + threadIdx.x;
  if (g >= G) return;
  float inv = 1.0f / fmaxf(pcnt[g], 1.0f);
  float acc = bf2[0];
  for (int j = 0; j < 32; ++j) {
    float q = bf1[j];
    for (int k = 0; k < 64; ++k) q += psum[g * HIDF + k] * inv * Wf1[k * 32 + j];
    acc += fmaxf(q, 0.0f) * Wf2[j];
  }
  out[g] = acc;
}

// ---------------------------------------------------------------------------
static inline int cdiv(long long a, long long b) { return (int)((a + b - 1) / b); }

extern "C" void kernel_launch(void* const* d_in, const int* in_sizes, int n_in,
                              void* d_out, int out_size, void* d_ws, size_t ws_size,
                              hipStream_t stream) {
  (void)n_in; (void)ws_size;
  const float* x    = (const float*)d_in[0];
  const int*   ei   = (const int*)d_in[1];
  const int*   bat  = (const int*)d_in[3];
  const float *W1  = (const float*)d_in[4],  *b1  = (const float*)d_in[5];
  const float *W2  = (const float*)d_in[6],  *b2  = (const float*)d_in[7];
  const float *Wg1 = (const float*)d_in[8],  *as1 = (const float*)d_in[9];
  const float *ad1 = (const float*)d_in[10], *bg1 = (const float*)d_in[11];
  const float *Wg2 = (const float*)d_in[12], *as2 = (const float*)d_in[13];
  const float *ad2 = (const float*)d_in[14], *bg2 = (const float*)d_in[15];
  const float *Wf1 = (const float*)d_in[16], *bf1 = (const float*)d_in[17];
  const float *Wf2 = (const float*)d_in[18], *bf2 = (const float*)d_in[19];

  const int INF = 32;
  const int N = in_sizes[0] / INF;   // 50000
  const int E = in_sizes[1] / 2;     // 800000
  const int G = out_size;            // 64
  const int* src = ei;
  const int* dst = ei + E;
  float* out = (float*)d_out;

  // Carve workspace (256B-aligned slices)
  char* wp = (char*)d_ws;
  auto carve = [&](size_t bytes) {
    char* r = wp;
    wp += (bytes + 255) & ~(size_t)255;
    return (void*)r;
  };
  float*    buf0  = (float*)carve((size_t)N * HIDF * 4);
  float*    buf1  = (float*)carve((size_t)N * HIDF * 4);
  float*    buf2  = (float*)carve((size_t)N * HIDF * 4);
  float*    dinv  = (float*)carve((size_t)N * 4);
  float*    es    = (float*)carve((size_t)N * 4);
  float*    ed    = (float*)carve((size_t)N * 4);
  float*    denom = (float*)carve((size_t)N * 4);
  float*    esel  = (float*)carve((size_t)N * 4);
  unsigned* emax  = (unsigned*)carve((size_t)N * 4);
  float*    ebuf  = (float*)carve((size_t)E * 4);
  float*    psum  = (float*)carve((size_t)G * HIDF * 4);
  float*    pcnt  = (float*)carve((size_t)G * 4);

  const int NT = 256;
  const int gN   = cdiv(N, NT);
  const int gNF  = cdiv((long long)N * HIDF, NT);
  const int gE   = cdiv(E, NT);
  const int gE32 = cdiv((long long)E * 32, NT);
  const int gN32 = cdiv((long long)N * 32, NT);
  const int gMM  = cdiv(N, 128);  // 8 waves/block, 16 rows/wave

  // --- degree / symmetric normalization (self loop => deg init 1) ---
  fill_f32<<<gN, NT, 0, stream>>>(dinv, 1.0f, N);
  deg_count<<<gE, NT, 0, stream>>>(dst, dinv, E);
  to_rsqrt<<<gN, NT, 0, stream>>>(dinv, N);

  // --- GCN layer helper ---
  auto run_gcn = [&](const float* Ain, int K, const float* Wt, const float* bias,
                     float* T, float* O) {
    gemm_wmma_bf16<<<gMM, NT, 0, stream>>>(Ain, Wt, T, N, K);
    fill_f32<<<gNF, NT, 0, stream>>>(O, 0.0f, N * HIDF);
    gcn_scatter<<<gE32, NT, 0, stream>>>(src, dst, dinv, T, O, E);
    gcn_finalize<<<gNF, NT, 0, stream>>>(O, T, dinv, bias, N);
  };
  // --- GAT layer helper ---
  auto run_gat = [&](const float* Ain, const float* Wt, const float* a_s,
                     const float* a_d, const float* bias, float* T, float* O) {
    gemm_wmma_bf16<<<gMM, NT, 0, stream>>>(Ain, Wt, T, N, HIDF);
    gat_scores<<<gN32, NT, 0, stream>>>(T, a_s, a_d, es, ed, N);
    fill_u32<<<gN, NT, 0, stream>>>(emax, NEG_INF_ENC, N);
    fill_f32<<<gN, NT, 0, stream>>>(denom, 0.0f, N);
    fill_f32<<<gNF, NT, 0, stream>>>(O, 0.0f, N * HIDF);
    gat_edge_max<<<gE, NT, 0, stream>>>(src, dst, es, ed, ebuf, emax, E);
    gat_self_max<<<gN, NT, 0, stream>>>(es, ed, esel, emax, N);
    gat_edge_exp<<<gE, NT, 0, stream>>>(dst, ebuf, emax, denom, E);
    gat_self_exp<<<gN, NT, 0, stream>>>(esel, emax, denom, N);
    gat_scatter<<<gE32, NT, 0, stream>>>(src, dst, ebuf, denom, T, O, E);
    gat_finalize<<<gNF, NT, 0, stream>>>(O, T, esel, denom, bias, N);
  };

  run_gcn(x,    INF,  W1, b1, buf0, buf1);        // h1 = relu(GCN(x))
  run_gcn(buf1, HIDF, W2, b2, buf0, buf2);        // h2 = relu(GCN(h1))
  run_gat(buf2, Wg1, as1, ad1, bg1, buf0, buf1);  // h3 = relu(GAT(h2))
  run_gat(buf1, Wg2, as2, ad2, bg2, buf0, buf2);  // h4 = relu(GAT(h3))

  // --- mean pool + MLP head ---
  fill_f32<<<cdiv((long long)G * HIDF, NT), NT, 0, stream>>>(psum, 0.0f, G * HIDF);
  fill_f32<<<1, NT, 0, stream>>>(pcnt, 0.0f, G);
  pool_sum<<<gN32, NT, 0, stream>>>(buf2, bat, psum, pcnt, N);
  mlp_head<<<1, 64, 0, stream>>>(psum, pcnt, Wf1, bf1, Wf2, bf2, out, G);
}
